// ClassifierGuided_3100966387979
// MI455X (gfx1250) — matrix operand order repaired and checked by reading, not compiled, not run on hardware
//
#include <hip/hip_runtime.h>
#include <hip/hip_bf16.h>

typedef __attribute__((ext_vector_type(16))) _Float16 v16h;
typedef __attribute__((ext_vector_type(8)))  _Float16 v8h;
typedef __attribute__((ext_vector_type(8)))  float    v8f;

#define TOKENS 4096
#define DMODEL 768
#define NEXP   16
#define DHID   192
#define HCAT   (NEXP*DHID)   /* 3072 */
#define NCLS   101

// ---------------------------------------------------------------------------
// CDNA5 async memory->LDS copy (16B per lane), tracked by ASYNCcnt.
// ---------------------------------------------------------------------------
__device__ __forceinline__ void async_copy_b128(const _Float16* gptr, _Float16* lptr) {
    unsigned lds = (unsigned)(unsigned long long)(void*)lptr;   // low 32b = LDS offset
    asm volatile("global_load_async_to_lds_b128 %0, %1, off"
                 :: "v"(lds), "v"(gptr) : "memory");
}
__device__ __forceinline__ void wait_async_le4() {
    asm volatile("s_wait_asynccnt 0x4" ::: "memory");
}
__device__ __forceinline__ void wait_async_0() {
    asm volatile("s_wait_asynccnt 0x0" ::: "memory");
}

// ---------------------------------------------------------------------------
// Small prep kernels
// ---------------------------------------------------------------------------
__global__ void cvt_f32_to_f16(const float* __restrict__ in, _Float16* __restrict__ out, int n) {
    int i = blockIdx.x * 256 + threadIdx.x;
    if (i < n) out[i] = (_Float16)in[i];
}

// W1: [16][768][192] -> W1T: [3072][768]  (row c=e*192+j holds K-dim contiguous)
__global__ void tr_w1_f16(const float* __restrict__ W1, _Float16* __restrict__ out) {
    int i = blockIdx.x * 256 + threadIdx.x;          // over 3072*768
    int c = i / DMODEL, k = i % DMODEL;
    int e = c / DHID,   j = c % DHID;
    out[i] = (_Float16)W1[(size_t)e * DMODEL * DHID + (size_t)k * DHID + j];
}

// W2cat: [3072][768] row-major -> W2T: [768][3072]
__global__ void tr_w2_f16(const float* __restrict__ W2, _Float16* __restrict__ out) {
    int i = blockIdx.x * 256 + threadIdx.x;          // over 768*3072
    int o = i / HCAT, k = i % HCAT;
    out[i] = (_Float16)W2[(size_t)k * DMODEL + o];
}

// Wo: [768][101] -> WoT: [128][768], zero padded cols 101..127
__global__ void tr_wo_f16(const float* __restrict__ Wo, _Float16* __restrict__ out) {
    int i = blockIdx.x * 256 + threadIdx.x;          // over 128*768
    int o = i / DMODEL, d = i % DMODEL;
    out[i] = (o < NCLS) ? (_Float16)Wo[(size_t)d * NCLS + o] : (_Float16)0.f;
}

// logits[b][e] = x[b] . Wg[:,e]   (fp32, matches reference gate precision)
__global__ void gate_logits(const float* __restrict__ xm, const float* __restrict__ Wg,
                            float* __restrict__ logits) {
    int t = blockIdx.x * 256 + threadIdx.x;          // 4096*16
    int b = t >> 4, e = t & 15;
    const float* xr = xm + (size_t)b * DMODEL;
    float a = 0.f;
    for (int d = 0; d < DMODEL; ++d) a = fmaf(xr[d], Wg[d * NEXP + e], a);
    logits[t] = a;
}

// top-12-of-16 softmax gates (drop the 4 smallest logits, softmax over the rest)
__global__ void gate_topk_softmax(const float* __restrict__ logits, float* __restrict__ gates) {
    int b = blockIdx.x * 256 + threadIdx.x;
    if (b >= TOKENS) return;
    float l[16]; bool keep[16];
    for (int i = 0; i < 16; ++i) { l[i] = logits[b * 16 + i]; keep[i] = true; }
    for (int it = 0; it < 4; ++it) {
        int mi = 0; float mv = 3.4e38f;
        for (int i = 0; i < 16; ++i) if (keep[i] && l[i] < mv) { mv = l[i]; mi = i; }
        keep[mi] = false;
    }
    float mx = -3.4e38f;
    for (int i = 0; i < 16; ++i) if (keep[i] && l[i] > mx) mx = l[i];
    float s = 0.f, ev[16];
    for (int i = 0; i < 16; ++i) { ev[i] = keep[i] ? expf(l[i] - mx) : 0.f; s += ev[i]; }
    float inv = 1.f / s;
    for (int i = 0; i < 16; ++i) gates[b * 16 + i] = ev[i] * inv;
}

// gb2[b][o] = sum_e gates[b][e] * b2[e][o]
__global__ void gate_bias2(const float* __restrict__ gates, const float* __restrict__ b2,
                           float* __restrict__ gb2) {
    int i = blockIdx.x * 256 + threadIdx.x;          // 4096*768
    int b = i / DMODEL, o = i % DMODEL;
    float a = 0.f;
    for (int e = 0; e < NEXP; ++e) a = fmaf(gates[b * 16 + e], b2[e * DMODEL + o], a);
    gb2[i] = a;
}

// ---------------------------------------------------------------------------
// WMMA GEMM: C[4096 x N] = A[4096 x K] * B[K x N], B supplied transposed [N][K].
// 256 threads = 8 waves (2x4), block tile 128x128, wave tile 64x32, BK=32.
// Global->LDS via async copies (ASYNCcnt), triple-buffered, 1 barrier/K-step.
// EPI 1: hg  = relu(C + b1[col]) * gates[row, col/192]          (f16 out)
// EPI 2: z   = relu(C + gb2[row,col]) + x[row,col]              (f16 out)
// EPI 3: out = C + bo[col], guarded col < 101                   (f32 out)
// ---------------------------------------------------------------------------
template<int EPI>
__global__ __launch_bounds__(256) void gemm_wmma_f16(
    const _Float16* __restrict__ A, const _Float16* __restrict__ BT, int K,
    const float* __restrict__ aux0, const float* __restrict__ aux1,
    void* __restrict__ outp)
{
    constexpr int BM = 128, BN = 128, BK = 32, LDT = BK + 8;   // LDT in halves
    __shared__ _Float16 As[3][BM * LDT];
    __shared__ _Float16 Bs[3][BN * LDT];

    const int tid  = threadIdx.x;
    const int lane = tid & 31;
    const int wave = tid >> 5;
    const int wm = wave >> 2, wn = wave & 3;       // 2 x 4 wave grid
    const int bm = blockIdx.x * BM;
    const int bn = blockIdx.y * BN;

    const int grow = tid >> 2;           // 0..63
    const int gseg = (tid & 3) * 8;      // half offset: 0,8,16,24

    const _Float16* Ab = A  + (size_t)bm * K;
    const _Float16* Bb = BT + (size_t)bn * K;

    auto issue = [&](int k0, int buf) {
        async_copy_b128(Ab + (size_t)grow        * K + k0 + gseg, &As[buf][grow * LDT + gseg]);
        async_copy_b128(Ab + (size_t)(grow + 64) * K + k0 + gseg, &As[buf][(grow + 64) * LDT + gseg]);
        async_copy_b128(Bb + (size_t)grow        * K + k0 + gseg, &Bs[buf][grow * LDT + gseg]);
        async_copy_b128(Bb + (size_t)(grow + 64) * K + k0 + gseg, &Bs[buf][(grow + 64) * LDT + gseg]);
    };

    v8f acc[4][2] = {};

    const int kb = ((lane >> 4) & 1) * 16;   // B frag K offset (halves)
    const int ka = ((lane >> 4) & 1) * 16;   // A frag K offset (halves)

    auto step = [&](int buf) {
        // ---- B fragments: lane holds col N, K contiguous;
        //      lanes 0-15 -> K 0..15, lanes 16-31 -> K 16..31  (contiguous order, same as A)
        v16h bf[2];
        #pragma unroll
        for (int u = 0; u < 2; ++u) {
            const int nb = wn * 32 + u * 16 + (lane & 15);
            const _Float16* bp = &Bs[buf][nb * LDT + kb];
            v8h blo = *(const v8h*)(bp);
            v8h bhi = *(const v8h*)(bp + 8);
            bf[u] = __builtin_shufflevector(blo, bhi, 0,1,2,3,4,5,6,7,8,9,10,11,12,13,14,15);
        }
        // ---- A fragments: lane<16 -> K 0..15 ; lane>=16 -> K 16..31
        #pragma unroll
        for (int t = 0; t < 4; ++t) {
            const int r = wm * 64 + t * 16 + (lane & 15);
            const _Float16* ap = &As[buf][r * LDT + ka];
            v8h alo = *(const v8h*)(ap);
            v8h ahi = *(const v8h*)(ap + 8);
            v16h af = __builtin_shufflevector(alo, ahi, 0,1,2,3,4,5,6,7,8,9,10,11,12,13,14,15);
            #pragma unroll
            for (int u = 0; u < 2; ++u) {
                acc[t][u] = __builtin_amdgcn_wmma_f32_16x16x32_f16(
                                false, af, false, bf[u], (short)0, acc[t][u], false, false);
            }
        }
    };

    const int nk = K / BK;
    issue(0, 0);
    int bufc = 0, bufn = 1;
    // Steady state: unconditional issue -> wait(<=4) -> barrier -> compute.
    for (int kt = 0; kt + 1 < nk; ++kt) {
        issue((kt + 1) * BK, bufn);
        // speculative L2 prefetch two tiles ahead (OOB prefetch is dropped)
        __builtin_prefetch(Ab + (size_t)grow * K + (kt + 2) * BK + gseg, 0, 1);
        __builtin_prefetch(Bb + (size_t)grow * K + (kt + 2) * BK + gseg, 0, 1);
        wait_async_le4();            // tile kt's 4 copies have landed
        __syncthreads();
        step(bufc);
        bufc = bufn;
        bufn = (bufn == 2) ? 0 : bufn + 1;
    }
    // Final K-step.
    wait_async_0();
    __syncthreads();
    step(bufc);

    // ---- Epilogue. C/D layout: VGPR i -> M = i (lanes 0-15) or i+8 (lanes 16-31)
    const int rowoff = (lane >> 4) << 3;
    #pragma unroll
    for (int t = 0; t < 4; ++t) {
        const int rbase = bm + wm * 64 + t * 16 + rowoff;
        #pragma unroll
        for (int u = 0; u < 2; ++u) {
            const int colb = bn + wn * 32 + u * 16 + (lane & 15);
            #pragma unroll
            for (int i = 0; i < 8; ++i) {
                const int row = rbase + i;
                const float c = acc[t][u][i];
                if (EPI == 1) {
                    float v = c + aux0[colb];
                    v = v > 0.f ? v : 0.f;
                    v *= aux1[row * NEXP + colb / DHID];
                    ((_Float16*)outp)[(size_t)row * HCAT + colb] = (_Float16)v;
                } else if (EPI == 2) {
                    const size_t idx = (size_t)row * DMODEL + colb;
                    float v = c + aux0[idx];
                    v = v > 0.f ? v : 0.f;
                    v += aux1[idx];
                    ((_Float16*)outp)[idx] = (_Float16)v;
                } else {
                    if (colb < NCLS)
                        ((float*)outp)[(size_t)row * NCLS + colb] = c + aux0[colb];
                }
            }
        }
    }
}

// ---------------------------------------------------------------------------
extern "C" void kernel_launch(void* const* d_in, const int* in_sizes, int n_in,
                              void* d_out, int out_size, void* d_ws, size_t ws_size,
                              hipStream_t stream)
{
    const float* xin[2] = { (const float*)d_in[0], (const float*)d_in[1] };
    const float* Wg = (const float*)d_in[2];
    const float* W1 = (const float*)d_in[3];
    const float* b1 = (const float*)d_in[4];
    const float* W2 = (const float*)d_in[5];
    const float* b2 = (const float*)d_in[6];
    const float* Wo = (const float*)d_in[7];
    const float* bo = (const float*)d_in[8];
    float* out = (float*)d_out;

    char* w = (char*)d_ws;
    auto carve = [&](size_t bytes) -> void* {
        void* p = (void*)w; w += (bytes + 255) & ~(size_t)255; return p;
    };
    _Float16* xh     = (_Float16*)carve((size_t)TOKENS * DMODEL * 2);
    _Float16* W1T    = (_Float16*)carve((size_t)HCAT   * DMODEL * 2);
    _Float16* W2T    = (_Float16*)carve((size_t)DMODEL * HCAT   * 2);
    _Float16* WoT    = (_Float16*)carve((size_t)128    * DMODEL * 2);
    _Float16* hg     = (_Float16*)carve((size_t)TOKENS * HCAT   * 2);
    _Float16* z      = (_Float16*)carve((size_t)TOKENS * DMODEL * 2);
    float*    logits = (float*)   carve((size_t)TOKENS * NEXP * 4);
    float*    gates  = (float*)   carve((size_t)TOKENS * NEXP * 4);
    float*    gb2    = (float*)   carve((size_t)TOKENS * DMODEL * 4);

    for (int m = 0; m < 2; ++m) {
        const float* xm = xin[m];
        cvt_f32_to_f16   <<<12288, 256, 0, stream>>>(xm, xh, TOKENS * DMODEL);
        tr_w1_f16        <<< 9216, 256, 0, stream>>>(W1 + (size_t)m * NEXP * DMODEL * DHID, W1T);
        tr_w2_f16        <<< 9216, 256, 0, stream>>>(W2 + (size_t)m * NEXP * DHID * DMODEL, W2T);
        tr_wo_f16        <<<  384, 256, 0, stream>>>(Wo + (size_t)m * DMODEL * NCLS, WoT);
        gate_logits      <<<  256, 256, 0, stream>>>(xm, Wg + (size_t)m * DMODEL * NEXP, logits);
        gate_topk_softmax<<<   16, 256, 0, stream>>>(logits, gates);
        gate_bias2       <<<12288, 256, 0, stream>>>(gates, b2 + (size_t)m * NEXP * DMODEL, gb2);

        // h_g = relu(x @ W1cat + b1) * gate       [4096 x 3072]
        gemm_wmma_f16<1><<<dim3(32, 24), 256, 0, stream>>>(
            xh, W1T, DMODEL, b1 + (size_t)m * HCAT, gates, (void*)hg);
        // z = relu(h_g @ W2cat + gb2) + x         [4096 x 768]
        gemm_wmma_f16<2><<<dim3(32, 6), 256, 0, stream>>>(
            hg, W2T, HCAT, gb2, xm, (void*)z);
        // out = z @ Wo + bo                       [4096 x 101]
        gemm_wmma_f16<3><<<dim3(32, 1), 256, 0, stream>>>(
            z, WoT, DMODEL, bo + (size_t)m * NCLS, (const float*)nullptr,
            (void*)(out + (size_t)m * TOKENS * NCLS));
    }
}
